// RaCo_35399120453938
// MI455X (gfx1250) — matrix operand order repaired and checked by reading, not compiled, not run on hardware
//
#include <hip/hip_runtime.h>
#include <hip/hip_bf16.h>
#include <stdint.h>

#define IMG_H   640
#define IMG_W   640
#define IMG_HW  (IMG_H * IMG_W)
#define BATCH   4
#define CHANS   64
#define KPTS    2048
#define OUTC    67          // x, y, 65 sampled channels
#define NBINS   4096
#define CAP     8192        // candidate capacity per batch (>= KPTS + tie slack)

#define TILE_W  64
#define TILE_H  32
#define HALO_W  (TILE_W + 2)
#define HALO_H  (TILE_H + 2)
#define HALO_P  68          // padded pitch (bank spread)

typedef __attribute__((ext_vector_type(16))) _Float16 v16h;
typedef __attribute__((ext_vector_type(8)))  float    v8f;

// ---------------- CDNA5 async global->LDS helpers -------------------------
__device__ __forceinline__ uint32_t lds_off(const void* p) {
  // generic shared pointer = {aperture_hi32, lds_offset32}; low 32 bits are
  // the LDS byte address.
  return (uint32_t)(uintptr_t)p;
}
__device__ __forceinline__ void async_ld_b32(uint32_t ldsaddr, const float* g) {
  // GV mode: vdst = LDS byte addr, vaddr = 64-bit global addr, saddr = off
  asm volatile("global_load_async_to_lds_b32 %0, %1, off"
               :: "v"(ldsaddr), "v"((unsigned long long)(uintptr_t)g)
               : "memory");
}
__device__ __forceinline__ void wait_async0() {
  asm volatile("s_wait_asynccnt 0" ::: "memory");
}

// ---------------- K0: zero histogram + counters ---------------------------
__global__ __launch_bounds__(256) void k_init(unsigned int* hist, int* count) {
  int i = blockIdx.x * blockDim.x + threadIdx.x;
  if (i < BATCH * NBINS) hist[i] = 0u;
  if (i < BATCH) count[i] = 0;
}

// ---------------- shared tile fill (async) --------------------------------
__device__ __forceinline__ void fill_tile(float (*tile)[HALO_P],
                                          const float* __restrict__ base,
                                          int ty0, int tx0, int tid) {
  for (int i = tid; i < HALO_H * HALO_W; i += 256) {
    int r = i / HALO_W, c = i % HALO_W;
    int gy = ty0 - 1 + r; gy = gy < 0 ? 0 : (gy > IMG_H - 1 ? IMG_H - 1 : gy);
    int gx = tx0 - 1 + c; gx = gx < 0 ? 0 : (gx > IMG_W - 1 ? IMG_W - 1 : gx);
    async_ld_b32(lds_off(&tile[r][c]), base + (size_t)gy * IMG_W + gx);
  }
  wait_async0();
  __syncthreads();
}

// ---------------- K1: NMS + per-batch value histogram ---------------------
__global__ __launch_bounds__(256) void k_nms_hist(const float* __restrict__ probs,
                                                  unsigned int* __restrict__ hist) {
  __shared__ float tile[HALO_H][HALO_P];
  __shared__ unsigned int lh[NBINS];
  const int tid = threadIdx.x;
  const int b   = blockIdx.z;
  const int tx0 = blockIdx.x * TILE_W;
  const int ty0 = blockIdx.y * TILE_H;
  const float* base = probs + (size_t)b * IMG_HW;

  for (int i = tid; i < NBINS; i += 256) lh[i] = 0u;
  fill_tile(tile, base, ty0, tx0, tid);

  for (int i = tid; i < TILE_H * TILE_W; i += 256) {
    int oy = i >> 6, ox = i & 63;
    int y = ty0 + oy, x = tx0 + ox;
    float p = tile[oy + 1][ox + 1];
    float m = p;
    #pragma unroll
    for (int dy = -1; dy <= 1; ++dy)
      #pragma unroll
      for (int dx = -1; dx <= 1; ++dx) {
        if (dy == 0 && dx == 0) continue;
        int yy = y + dy, xx = x + dx;
        if (yy < 0 || yy >= IMG_H || xx < 0 || xx >= IMG_W) continue; // -inf pad
        m = fmaxf(m, tile[oy + 1 + dy][ox + 1 + dx]);
      }
    if (p == m) {                      // NMS survivor
      int bin = (int)(p * (float)NBINS);
      bin = bin < 0 ? 0 : (bin > NBINS - 1 ? NBINS - 1 : bin);
      atomicAdd(&lh[bin], 1u);
    }
  }
  __syncthreads();
  for (int i = tid; i < NBINS; i += 256)
    if (lh[i]) atomicAdd(&hist[b * NBINS + i], lh[i]);
}

// ---------------- K2: per-batch threshold bin -----------------------------
__global__ void k_thresh(const unsigned int* __restrict__ hist, int* __restrict__ tbin) {
  int b = threadIdx.x;
  if (b < BATCH) {
    unsigned int cum = 0; int t = 0;
    for (int bin = NBINS - 1; bin >= 0; --bin) {
      cum += hist[b * NBINS + bin];
      if (cum >= KPTS) { t = bin; break; }
    }
    tbin[b] = t;
  }
}

// ---------------- K3: gather candidates >= threshold bin ------------------
__global__ __launch_bounds__(256) void k_gather(const float* __restrict__ probs,
                                                const int* __restrict__ tbin,
                                                int* __restrict__ count,
                                                unsigned long long* __restrict__ cand) {
  __shared__ float tile[HALO_H][HALO_P];
  const int tid = threadIdx.x;
  const int b   = blockIdx.z;
  const int tx0 = blockIdx.x * TILE_W;
  const int ty0 = blockIdx.y * TILE_H;
  const int tb  = tbin[b];
  const float* base = probs + (size_t)b * IMG_HW;

  fill_tile(tile, base, ty0, tx0, tid);

  for (int i = tid; i < TILE_H * TILE_W; i += 256) {
    int oy = i >> 6, ox = i & 63;
    int y = ty0 + oy, x = tx0 + ox;
    float p = tile[oy + 1][ox + 1];
    float m = p;
    #pragma unroll
    for (int dy = -1; dy <= 1; ++dy)
      #pragma unroll
      for (int dx = -1; dx <= 1; ++dx) {
        if (dy == 0 && dx == 0) continue;
        int yy = y + dy, xx = x + dx;
        if (yy < 0 || yy >= IMG_H || xx < 0 || xx >= IMG_W) continue;
        m = fmaxf(m, tile[oy + 1 + dy][ox + 1 + dx]);
      }
    if (p == m) {
      int bin = (int)(p * (float)NBINS);
      bin = bin < 0 ? 0 : (bin > NBINS - 1 ? NBINS - 1 : bin);
      if (bin >= tb) {
        unsigned int idx = (unsigned int)(y * IMG_W + x);
        // value desc, then index asc (matches jax.lax.top_k tie order)
        unsigned long long key =
            ((unsigned long long)__float_as_uint(p) << 32) | (unsigned int)(~idx);
        int pos = atomicAdd(&count[b], 1);
        if (pos < CAP) cand[(size_t)b * CAP + pos] = key;
      }
    }
  }
}

// ---------------- K4: exact top-K via 8192-wide bitonic sort in LDS -------
__global__ __launch_bounds__(1024) void k_sort(const unsigned long long* __restrict__ cand,
                                               const int* __restrict__ count,
                                               unsigned int* __restrict__ kp_idx) {
  __shared__ unsigned long long s[CAP];   // 64 KB LDS
  const int b = blockIdx.x, tid = threadIdx.x;
  int n = count[b]; if (n > CAP) n = CAP;
  for (int i = tid; i < CAP; i += 1024)
    s[i] = (i < n) ? cand[(size_t)b * CAP + i] : 0ull;   // 0 sorts last
  __syncthreads();
  for (int k = 2; k <= CAP; k <<= 1) {
    for (int j = k >> 1; j > 0; j >>= 1) {
      for (int i = tid; i < CAP; i += 1024) {
        int ij = i ^ j;
        if (ij > i) {
          unsigned long long a = s[i], c = s[ij];
          bool descBlock = ((i & k) == 0);            // overall descending
          if (descBlock ? (a < c) : (a > c)) { s[i] = c; s[ij] = a; }
        }
      }
      __syncthreads();
    }
  }
  for (int i = tid; i < KPTS; i += 1024) {
    unsigned int idx = ~(unsigned int)(s[i] & 0xffffffffu);
    if (idx >= IMG_HW) idx = 0u;
    kp_idx[b * KPTS + i] = idx;
  }
}

// ---------------- K5: 3x3 softmax + WMMA offsets einsum -------------------
// One wave handles 16 keypoints: D[16x16] = A(16x32 f16 softmax wts) x B(32x16 offsets)
__global__ __launch_bounds__(256) void k_subpix(const float* __restrict__ logits,
                                                const unsigned int* __restrict__ kp_idx,
                                                float* __restrict__ out,
                                                float* __restrict__ kxy) {
  const int gt   = blockIdx.x * blockDim.x + threadIdx.x;
  const int wave = gt >> 5;
  const int lane = gt & 31;
  const int b    = wave / (KPTS / 16);
  const int m0   = (wave % (KPTS / 16)) * 16;
  const int m    = lane & 15;

  const unsigned int idx = kp_idx[b * KPTS + m0 + m];
  const int h = (int)(idx / IMG_W), w = (int)(idx % IMG_W);
  const float* lg = logits + (size_t)b * IMG_HW;

  float t[9];
  #pragma unroll
  for (int k = 0; k < 9; ++k) {
    int dy = k / 3 - 1, dx = k % 3 - 1;
    int yy = h + dy, xx = w + dx;
    bool inb = (yy >= 0 && yy < IMG_H && xx >= 0 && xx < IMG_W);
    float v = inb ? lg[(size_t)yy * IMG_W + xx] : 0.0f;  // zero-padded logits
    t[k] = v * 2.0f;                                      // / SUBPIXEL_TEMP
  }
  float mx = t[0];
  #pragma unroll
  for (int k = 1; k < 9; ++k) mx = fmaxf(mx, t[k]);
  float e[9]; float sum = 0.0f;
  #pragma unroll
  for (int k = 0; k < 9; ++k) { e[k] = __expf(t[k] - mx); sum += e[k]; }
  const float inv = 1.0f / sum;

  // A fragment: lanes 0-15 carry row M=lane's 9 weights in elements 0..8; lanes 16-31 are zero
  v16h a;
  #pragma unroll
  for (int i = 0; i < 16; ++i) a[i] = (_Float16)0.0f;
  if (lane < 16) {
    #pragma unroll
    for (int k = 0; k < 9; ++k) a[k] = (_Float16)(e[k] * inv);
  }
  // B fragment: lane N (0-15) holds column N, taps in elements 0..8 (same order as A); col0 = gx, col1 = gy
  v16h bm;
  #pragma unroll
  for (int i = 0; i < 16; ++i) bm[i] = (_Float16)0.0f;
  if (lane == 0) {
    #pragma unroll
    for (int k = 0; k < 9; ++k) bm[k] = (_Float16)(float)(k % 3 - 1);
  }
  if (lane == 1) {
    #pragma unroll
    for (int k = 0; k < 9; ++k) bm[k] = (_Float16)(float)(k / 3 - 1);
  }
  v8f c;
  #pragma unroll
  for (int i = 0; i < 8; ++i) c[i] = 0.0f;
  c = __builtin_amdgcn_wmma_f32_16x16x32_f16(false, a, false, bm,
                                             (short)0, c, false, false);

  // D layout: lane n holds D[M=0..7][n] in c[0..7]; lane n+16 holds M=8..15
  const bool isx = (lane == 0) || (lane == 16);
  const bool isy = (lane == 1) || (lane == 17);
  if (isx || isy) {
    const int mbase = (lane >= 16) ? 8 : 0;
    #pragma unroll
    for (int j = 0; j < 8; ++j) {
      const int mm = mbase + j;
      const unsigned int id2 = kp_idx[b * KPTS + m0 + mm];
      const int hh = (int)(id2 / IMG_W), ww = (int)(id2 % IMG_W);
      const size_t g = (size_t)b * KPTS + m0 + mm;
      const float d = c[j];
      if (isx) { float xv = (float)ww + d; out[g * OUTC + 0] = xv; kxy[g * 2 + 0] = xv; }
      else     { float yv = (float)hh + d; out[g * OUTC + 1] = yv; kxy[g * 2 + 1] = yv; }
    }
  }
}

// ---------------- K6: bilinear gather of 65-channel stack -----------------
__global__ __launch_bounds__(256) void k_sample(const float* __restrict__ probs,
                                                const float* __restrict__ fm,
                                                const float* __restrict__ kxy,
                                                float* __restrict__ out) {
  const int gt   = blockIdx.x * blockDim.x + threadIdx.x;
  const int wave = gt >> 5;
  const int lane = gt & 31;
  const int b    = wave / KPTS;
  const int n    = wave % KPTS;
  const size_t g = (size_t)b * KPTS + n;

  float x = kxy[g * 2 + 0], y = kxy[g * 2 + 1];
  x = fminf(fmaxf(x, 0.0f), (float)(IMG_W - 1));
  y = fminf(fmaxf(y, 0.0f), (float)(IMG_H - 1));
  const float x0f = floorf(x), y0f = floorf(y);
  const float wx = x - x0f, wy = y - y0f;
  const int x0 = (int)x0f, y0 = (int)y0f;
  const int x1 = min(x0 + 1, IMG_W - 1), y1 = min(y0 + 1, IMG_H - 1);
  const float w00 = (1.f - wx) * (1.f - wy), w01 = wx * (1.f - wy);
  const float w10 = (1.f - wx) * wy,         w11 = wx * wy;
  const size_t i00 = (size_t)y0 * IMG_W + x0, i01 = (size_t)y0 * IMG_W + x1;
  const size_t i10 = (size_t)y1 * IMG_W + x0, i11 = (size_t)y1 * IMG_W + x1;

  // prefetch the second channel plane this lane will touch (global_prefetch_b8)
  __builtin_prefetch(fm + ((size_t)b * CHANS + (lane + 31)) * IMG_HW + i00, 0, 1);

  for (int ch = lane; ch < CHANS + 1; ch += 32) {
    const float* p = (ch == 0) ? (probs + (size_t)b * IMG_HW)
                               : (fm + ((size_t)b * CHANS + (ch - 1)) * IMG_HW);
    float v = p[i00] * w00 + p[i01] * w01 + p[i10] * w10 + p[i11] * w11;
    out[g * OUTC + 2 + ch] = v;
  }
}

// ---------------- host-side launch ----------------------------------------
extern "C" void kernel_launch(void* const* d_in, const int* in_sizes, int n_in,
                              void* d_out, int out_size, void* d_ws, size_t ws_size,
                              hipStream_t stream) {
  (void)in_sizes; (void)n_in; (void)out_size; (void)ws_size;
  const float* probs  = (const float*)d_in[0];   // [4,1,640,640]
  const float* logits = (const float*)d_in[1];   // [4,1,640,640]
  const float* fm     = (const float*)d_in[2];   // [4,64,640,640]
  float* out = (float*)d_out;                    // [4,2048,67]

  char* ws = (char*)d_ws;
  unsigned int*       hist   = (unsigned int*)(ws);                          // 64 KB
  int*                tbin   = (int*)(ws + 65536);                           // 16 B
  int*                count  = (int*)(ws + 65536 + 64);                      // 16 B
  unsigned long long* cand   = (unsigned long long*)(ws + 65536 + 128);      // 256 KB
  unsigned int*       kp_idx = (unsigned int*)(ws + 65536 + 128 + (size_t)BATCH * CAP * 8);
  float*              kxy    = (float*)(ws + 65536 + 128 + (size_t)BATCH * CAP * 8
                                          + (size_t)BATCH * KPTS * 4);

  const dim3 tgrid(IMG_W / TILE_W, IMG_H / TILE_H, BATCH);   // 10 x 20 x 4

  k_init   <<<(BATCH * NBINS + 255) / 256, 256, 0, stream>>>(hist, count);
  k_nms_hist<<<tgrid, 256, 0, stream>>>(probs, hist);
  k_thresh <<<1, 32, 0, stream>>>(hist, tbin);
  k_gather <<<tgrid, 256, 0, stream>>>(probs, tbin, count, cand);
  k_sort   <<<BATCH, 1024, 0, stream>>>(cand, count, kp_idx);
  k_subpix <<<(BATCH * KPTS / 16) * 32 / 256, 256, 0, stream>>>(logits, kp_idx, out, kxy);
  k_sample <<<(BATCH * KPTS) * 32 / 256, 256, 0, stream>>>(probs, fm, kxy, out);
}
